// enhance_net_nopool_91164975824969
// MI455X (gfx1250) — matrix operand order repaired and checked by reading, not compiled, not run on hardware
//
#include <hip/hip_runtime.h>
#include <hip/hip_bf16.h>
#include <math.h>

// ---------------------------------------------------------------------------
// MI455X (gfx1250) implementation of the Zero-DCE-like enhancement net.
//
// conv2/conv3 (32ch->32ch 3x3 over 8x512x512) carry ~155 of ~176 GFLOP ->
// implicit GEMM with V_WMMA_F32_16X16X32_F16 (f16 in, f32 acc). M = 16
// pixels, N = 16 output channels, K = 32 input channels per tap; 9 taps = 9
// chained WMMAs per tile. Intermediates are NHWC f16 (one pixel's channels =
// contiguous 64B). Halo tiles are staged into LDS with ASYNCcnt-tracked
// GLOBAL_LOAD_ASYNC_TO_LDS_B128 (guarded on builtin availability), otherwise
// b128 load/ds_store. Epilogue stages results through LDS so the global
// write-back is one coalesced b128 store per thread instead of 8 scattered
// b16 stores per lane. Mean-feedback iterations are pure bandwidth
// (~75 MB/iter ~ 3us at 23.3 TB/s) with deterministic 2-stage reductions and
// an on-device data-dependent trip count (graph-capture safe).
// ---------------------------------------------------------------------------

typedef __attribute__((ext_vector_type(16))) _Float16 v16h;
typedef __attribute__((ext_vector_type(8)))  float    v8f;

#define NPIX   262144      // 512*512 pixels per image
#define NTOT   6291456     // 8*3*512*512 elements of x
#define HH     512
#define WW     512

#if defined(__has_builtin)
#if __has_builtin(__builtin_amdgcn_global_load_async_to_lds_b128) && \
    __has_builtin(__builtin_amdgcn_s_wait_asynccnt)
#define USE_ASYNC_LDS 1
#endif
#endif
#ifndef USE_ASYNC_LDS
#define USE_ASYNC_LDS 0
#endif

#if USE_ASYNC_LDS
// Builtin signature wants NATIVE clang vector pointers (not HIP_vector_type):
//   (v4i32 addrspace(1)*, v4i32 addrspace(3)*, imm offset, imm cpol)
typedef int v4i32 __attribute__((__vector_size__(16)));
typedef __attribute__((address_space(1))) v4i32 as1_v4i32;
typedef __attribute__((address_space(3))) v4i32 as3_v4i32;
// 16B global -> LDS, no VGPR round trip; tracked by ASYNCcnt.
static __device__ __forceinline__ void async_cp16(const void* g, void* l) {
  __builtin_amdgcn_global_load_async_to_lds_b128(
      (as1_v4i32*)g, (as3_v4i32*)l, /*offset=*/0, /*cpol=*/0);
}
#endif

static __device__ __forceinline__ v8f wmma16(v16h a, v16h b, v8f c) {
  // D = A(16x32 f16) * B(32x16 f16) + C(16x16 f32)
  return __builtin_amdgcn_wmma_f32_16x16x32_f16(
      /*neg_a=*/false, a, /*neg_b=*/false, b,
      /*c_mod=*/(short)0, c, /*reuse_a=*/false, /*reuse_b=*/false);
}

// ---------------- weight repack: OIHW fp32 -> [tap][co][ci] f16 -------------
__global__ __launch_bounds__(256) void prep_w32(const float* __restrict__ w,
                                                _Float16* __restrict__ o) {
  int i = blockIdx.x * 256 + threadIdx.x;       // 9*32*32 = 9216
  if (i >= 9216) return;
  int tap = i / 1024, co = (i / 32) % 32, ci = i % 32;
  o[i] = (_Float16)w[(co * 32 + ci) * 9 + tap];
}

__global__ __launch_bounds__(256) void prep_w7(const float* __restrict__ w,
                                               _Float16* __restrict__ o) {
  int i = blockIdx.x * 256 + threadIdx.x;       // 9*16*64 = 9216
  if (i >= 9216) return;
  int tap = i / 1024, co = (i / 64) % 16, ci = i % 64;
  o[i] = (co < 3) ? (_Float16)w[(co * 64 + ci) * 9 + tap] : (_Float16)0.0f;
}

// ---------------- conv1: 3->32 direct (small), NCHW f32 -> NHWC f16 ---------
__global__ __launch_bounds__(256) void conv1_direct(
    const float* __restrict__ x, int img, int branch,
    const float* __restrict__ w1, const float* __restrict__ b1,
    _Float16* __restrict__ out) {
  __shared__ float sw[32 * 27];
  __shared__ float sb[32];
  int tid = threadIdx.x;
  for (int i = tid; i < 32 * 27; i += 256) sw[i] = w1[i];
  if (tid < 32) sb[tid] = b1[tid];
  __syncthreads();

  int p = blockIdx.x * 256 + tid;               // 0..262143
  int h = p >> 9, w = p & 511;
  const float* xb = x + (size_t)img * 3 * NPIX;

  float inv[27];
#pragma unroll
  for (int kh = 0; kh < 3; ++kh)
#pragma unroll
    for (int kw = 0; kw < 3; ++kw) {
      int gh = h + kh - 1, gw = w + kw - 1;
      bool ok = (gh >= 0) & (gh < HH) & (gw >= 0) & (gw < WW);
#pragma unroll
      for (int ci = 0; ci < 3; ++ci) {
        float v = 0.0f;
        if (ok) { v = xb[(ci * HH + gh) * WW + gw]; if (branch) v = 1.0f - v; }
        inv[ci * 9 + kh * 3 + kw] = v;
      }
    }
  _Float16* op = out + (size_t)p * 32;          // 64B contiguous per thread
  for (int co = 0; co < 32; ++co) {
    float a = sb[co];
    const float* wp = &sw[co * 27];
#pragma unroll
    for (int k = 0; k < 27; ++k) a = fmaf(inv[k], wp[k], a);
    op[co] = (_Float16)(a > 0.0f ? a : 0.0f);
  }
}

// ---------------- conv 32->32 3x3 via WMMA (ReLU), NHWC f16 -----------------
// Block: 256 thr = 8 waves; tile = 64 pixels (one row segment) x 32 co.
// Wave (q,t): q=wave&3 pixel subtile (16 px), t=wave>>2 co subtile (16 co).
__global__ __launch_bounds__(256) void conv3x3_c32(
    const _Float16* __restrict__ in,    // image NHWC f16
    const _Float16* __restrict__ wts,   // [9][32][32] f16
    const float* __restrict__ bias,     // 32 f32
    _Float16* __restrict__ out) {
  __shared__ alignas(16) _Float16 sIn[3 * 66 * 32];   // halo tile, 12672 B
  __shared__ alignas(16) _Float16 sW[9 * 32 * 32];    // 18432 B

  const int tid = threadIdx.x;
  const int h  = blockIdx.x >> 3;
  const int w0 = (blockIdx.x & 7) << 6;

  { // weights -> LDS (always L2-hot: 18 KB)
    const int4* s = (const int4*)wts;
    int4* d = (int4*)sW;
    for (int i = tid; i < 1152; i += 256) d[i] = s[i];
  }
  // input halo tile rows h-1..h+1, cols w0-1..w0+64 (zero padded)
  for (int e = tid; e < 3 * 66; e += 256) {
    int r = e / 66, c = e % 66;
    int gh = h + r - 1, gw = w0 + c - 1;
    _Float16* d = &sIn[(size_t)e * 32];
    if (gh >= 0 && gh < HH && gw >= 0 && gw < WW) {
      const _Float16* s = &in[((size_t)gh * WW + gw) * 32];
#if USE_ASYNC_LDS
      async_cp16((const char*)s,      (char*)d);
      async_cp16((const char*)s + 16, (char*)d + 16);
      async_cp16((const char*)s + 32, (char*)d + 32);
      async_cp16((const char*)s + 48, (char*)d + 48);
#else
      const int4* sq = (const int4*)s;
      int4* dq = (int4*)d;
      dq[0] = sq[0]; dq[1] = sq[1]; dq[2] = sq[2]; dq[3] = sq[3];
#endif
    } else {
      int4 z = {0, 0, 0, 0};
      int4* dq = (int4*)d;
      dq[0] = z; dq[1] = z; dq[2] = z; dq[3] = z;
    }
  }
#if USE_ASYNC_LDS
  __builtin_amdgcn_s_wait_asynccnt(0);   // own wave's async DMA done...
#endif
  __syncthreads();                       // ...then everyone's tile visible

  const int wave = tid >> 5, lane = tid & 31;
  const int p0  = (wave & 3) << 4;      // pixel subtile base
  const int cob = (wave >> 2) << 4;     // co subtile base
  const int lh  = lane & 15;
  const bool hi = lane >= 16;
  const int akb = hi ? 8 : 0;           // A: K halves {kb..kb+7, kb+16..kb+23}
  const int bkb = hi ? 8 : 0;           // B: K halves {kb..kb+7, kb+16..kb+23}

  v8f acc = {};
#pragma unroll
  for (int kh = 0; kh < 3; ++kh)
#pragma unroll
    for (int kw = 0; kw < 3; ++kw) {
      const int tap = kh * 3 + kw;
      const _Float16* ap = &sIn[((kh * 66) + (p0 + lh + kw)) * 32 + akb];
      const _Float16* bp = &sW[((tap * 32) + (cob + lh)) * 32 + bkb];
      v16h A, B;
      ((int4*)&A)[0] = *(const int4*)ap;         // K  0.. 7 (hi:  8..15)
      ((int4*)&A)[1] = *(const int4*)(ap + 16);  // K 16..23 (hi: 24..31)
      ((int4*)&B)[0] = *(const int4*)bp;         // K kb..kb+7
      ((int4*)&B)[1] = *(const int4*)(bp + 16);  // K kb+16..kb+23
      acc = wmma16(A, B, acc);
    }

  // epilogue: bias+ReLU -> stage f16 tile in LDS -> one b128 store per thread
  const int co = cob + lh;
  const float bv = bias[co];
  __syncthreads();                        // done reading sIn; reuse as sOut
  _Float16* sOut = sIn;                   // 64 px * 32 ch = 4096 B
#pragma unroll
  for (int r = 0; r < 8; ++r) {
    const int m = hi ? r + 8 : r;         // C/D layout: lanes>=16 -> M+8
    float v = acc[r] + bv;
    v = v > 0.0f ? v : 0.0f;
    sOut[(p0 + m) * 32 + co] = (_Float16)v;
  }
  __syncthreads();
  int4* gout = (int4*)(out + (((size_t)h * WW) + w0) * 32);  // 256 int4
  gout[tid] = ((const int4*)sOut)[tid];
}

// ---------------- conv7: concat(x1,x3)=64 -> 3 (padded to 16), tanh ---------
// Block: 128 thr = 4 waves; tile = 64 pixels x 16 co (3 real).
__global__ __launch_bounds__(128) void conv7_wmma(
    const _Float16* __restrict__ x1, const _Float16* __restrict__ x3,
    const _Float16* __restrict__ wts,   // [9][16][64] f16 (co padded)
    const float* __restrict__ b7,       // 3 f32
    float* __restrict__ xr,             // NCHW f32 (8,3,512,512)
    int img, int branch) {
  __shared__ alignas(16) _Float16 s1[3 * 66 * 32];
  __shared__ alignas(16) _Float16 s3[3 * 66 * 32];
  __shared__ alignas(16) _Float16 sW[9 * 16 * 64];

  const int tid = threadIdx.x;
  const int h  = blockIdx.x >> 3;
  const int w0 = (blockIdx.x & 7) << 6;

  {
    const int4* s = (const int4*)wts;
    int4* d = (int4*)sW;
    for (int i = tid; i < 1152; i += 128) d[i] = s[i];
  }
  for (int e = tid; e < 3 * 66; e += 128) {
    int r = e / 66, c = e % 66;
    int gh = h + r - 1, gw = w0 + c - 1;
    _Float16* d1 = &s1[(size_t)e * 32];
    _Float16* d3 = &s3[(size_t)e * 32];
    if (gh >= 0 && gh < HH && gw >= 0 && gw < WW) {
      const _Float16* a = &x1[((size_t)gh * WW + gw) * 32];
      const _Float16* b = &x3[((size_t)gh * WW + gw) * 32];
#if USE_ASYNC_LDS
      async_cp16((const char*)a,      (char*)d1);
      async_cp16((const char*)a + 16, (char*)d1 + 16);
      async_cp16((const char*)a + 32, (char*)d1 + 32);
      async_cp16((const char*)a + 48, (char*)d1 + 48);
      async_cp16((const char*)b,      (char*)d3);
      async_cp16((const char*)b + 16, (char*)d3 + 16);
      async_cp16((const char*)b + 32, (char*)d3 + 32);
      async_cp16((const char*)b + 48, (char*)d3 + 48);
#else
      const int4* aq = (const int4*)a;
      const int4* bq = (const int4*)b;
      int4* q1 = (int4*)d1;
      int4* q3 = (int4*)d3;
      q1[0] = aq[0]; q1[1] = aq[1]; q1[2] = aq[2]; q1[3] = aq[3];
      q3[0] = bq[0]; q3[1] = bq[1]; q3[2] = bq[2]; q3[3] = bq[3];
#endif
    } else {
      int4 z = {0, 0, 0, 0};
      int4* q1 = (int4*)d1;
      int4* q3 = (int4*)d3;
      q1[0] = z; q1[1] = z; q1[2] = z; q1[3] = z;
      q3[0] = z; q3[1] = z; q3[2] = z; q3[3] = z;
    }
  }
#if USE_ASYNC_LDS
  __builtin_amdgcn_s_wait_asynccnt(0);
#endif
  __syncthreads();

  const int wave = tid >> 5, lane = tid & 31;
  const int p0 = wave << 4;
  const int lh = lane & 15;
  const bool hi = lane >= 16;
  const int akb = hi ? 8 : 0;
  const int bkb = hi ? 8 : 0;

  v8f acc = {};
#pragma unroll
  for (int kh = 0; kh < 3; ++kh)
#pragma unroll
    for (int kw = 0; kw < 3; ++kw) {
      const int tap = kh * 3 + kw;
      const _Float16* ap1 = &s1[((kh * 66) + (p0 + lh + kw)) * 32 + akb];
      const _Float16* ap3 = &s3[((kh * 66) + (p0 + lh + kw)) * 32 + akb];
      const _Float16* bp  = &sW[((tap * 16) + lh) * 64 + bkb];
      v16h A1, A3, B1, B3;
      ((int4*)&A1)[0] = *(const int4*)ap1;
      ((int4*)&A1)[1] = *(const int4*)(ap1 + 16);
      ((int4*)&A3)[0] = *(const int4*)ap3;
      ((int4*)&A3)[1] = *(const int4*)(ap3 + 16);
      ((int4*)&B1)[0] = *(const int4*)bp;          // ci kb..kb+7, kb+16..kb+23 (x1 half)
      ((int4*)&B1)[1] = *(const int4*)(bp + 16);
      ((int4*)&B3)[0] = *(const int4*)(bp + 32);   // ci 32+kb..   (x3 half)
      ((int4*)&B3)[1] = *(const int4*)(bp + 48);
      acc = wmma16(A1, B1, acc);                   // K = 64 per tap -> 2 WMMAs
      acc = wmma16(A3, B3, acc);
    }

  const int co = lh;
  if (co < 3) {
    const float bv = b7[co];
#pragma unroll
    for (int r = 0; r < 8; ++r) {          // 32B contiguous per lane (f32)
      const int m = hi ? r + 8 : r;
      float v = 0.5f * tanhf(acc[r] + bv);
      size_t idx = (((size_t)img * 3 + co) * HH + h) * WW + (w0 + p0 + m);
      if (branch == 0) xr[idx] = v;     // branch(x)
      else             xr[idx] += v;    // + branch(1-x), averaged by the 0.5f
    }
  }
}

// ---------------- elementwise copy x -> out (f32) ---------------------------
__global__ __launch_bounds__(256) void copy_x(const float* __restrict__ src,
                                              float* __restrict__ dst) {
  size_t i = (size_t)blockIdx.x * 256 + threadIdx.x;   // float4 index
  ((float4*)dst)[i] = ((const float4*)src)[i];
}

// ---------------- two-stage deterministic mean reduction --------------------
__global__ __launch_bounds__(256) void reduce_partial(
    const float* __restrict__ src, float* __restrict__ part) {
  __shared__ float s[256];
  int tid = threadIdx.x;
  const float4* p = (const float4*)src + (size_t)blockIdx.x * 768;
  float a = 0.0f;
#pragma unroll
  for (int i = 0; i < 3; ++i) {
    float4 v = p[tid + i * 256];
    a += v.x + v.y + v.z + v.w;
  }
  s[tid] = a;
  __syncthreads();
  for (int o = 128; o > 0; o >>= 1) {
    if (tid < o) s[tid] += s[tid + o];
    __syncthreads();
  }
  if (tid == 0) part[blockIdx.x] = s[0];
}

__global__ __launch_bounds__(256) void reduce_final(
    const float* __restrict__ part, float* __restrict__ scal, int mode) {
  __shared__ float s[256];
  int tid = threadIdx.x;
  float a = 0.0f;
  for (int i = tid; i < 2048; i += 256) a += part[i];
  s[tid] = a;
  __syncthreads();
  for (int o = 128; o > 0; o >>= 1) {
    if (tid < o) s[tid] += s[tid + o];
    __syncthreads();
  }
  if (tid == 0) {
    float m = s[0] * (1.0f / (float)NTOT);
    if (mode == 0) {                      // initial stats: n3 and loop count b
      float sq = m * m;
      scal[0] = -0.79f * sq + 0.81f * m + 1.4f;        // n3
      float bf;
      if (m < 0.1f)       bf = -25.0f * m + 10.0f;
      else if (m < 0.45f) bf = 17.14f * sq - 15.14f * m + 10.0f;
      else                bf = 5.66f * sq - 2.93f * m + 7.2f;
      scal[1] = (float)(int)bf;                        // int() truncation
    }
    scal[2] = m;                                       // per-iteration mean
  }
}

// ---------------- curve iteration: x += xr*(x^2 - x)*((n1-m)/(n3-m)) --------
__global__ __launch_bounds__(256) void curve_update(
    float* __restrict__ xv, const float* __restrict__ xr,
    const float* __restrict__ scal, int it) {
  int b = (int)scal[1];
  if (it >= b) return;                 // data-dependent trip count, on-device
  float n3 = scal[0], m = scal[2];
  float f = (0.63f - m) / (n3 - m);
  size_t i = (size_t)blockIdx.x * 256 + threadIdx.x;
  float4 x = ((const float4*)xv)[i];
  float4 r = ((const float4*)xr)[i];
  x.x = fmaf(r.x * (x.x * x.x - x.x), f, x.x);
  x.y = fmaf(r.y * (x.y * x.y - x.y), f, x.y);
  x.z = fmaf(r.z * (x.z * x.z - x.z), f, x.z);
  x.w = fmaf(r.w * (x.w * x.w - x.w), f, x.w);
  ((float4*)xv)[i] = x;
}

// ---------------------------------------------------------------------------
extern "C" void kernel_launch(void* const* d_in, const int* in_sizes, int n_in,
                              void* d_out, int out_size, void* d_ws,
                              size_t ws_size, hipStream_t stream) {
  (void)in_sizes; (void)n_in; (void)out_size; (void)ws_size;
  const float* x  = (const float*)d_in[0];
  const float* w1 = (const float*)d_in[1];
  const float* b1 = (const float*)d_in[2];
  const float* w2 = (const float*)d_in[3];
  const float* b2 = (const float*)d_in[4];
  const float* w3 = (const float*)d_in[5];
  const float* b3 = (const float*)d_in[6];
  const float* w7 = (const float*)d_in[7];
  const float* b7 = (const float*)d_in[8];

  float* out_x  = (float*)d_out;
  float* out_xr = out_x + NTOT;

  char* ws = (char*)d_ws;                       // ~50.4 MB total
  _Float16* x1b = (_Float16*)(ws);              // 16,777,216 B (NHWC f16)
  _Float16* x2b = (_Float16*)(ws + 16777216);
  _Float16* x3b = (_Float16*)(ws + 33554432);
  _Float16* w2h = (_Float16*)(ws + 50331648);           // 18,432 B
  _Float16* w3h = (_Float16*)(ws + 50331648 + 18432);
  _Float16* w7h = (_Float16*)(ws + 50331648 + 36864);
  float* part = (float*)(ws + 50331648 + 55296);        // 2048 partial sums
  float* scal = part + 2048;                            // n3, b, m

  prep_w32<<<36, 256, 0, stream>>>(w2, w2h);
  prep_w32<<<36, 256, 0, stream>>>(w3, w3h);
  prep_w7 <<<36, 256, 0, stream>>>(w7, w7h);

  // two curve branches (x and 1-x), one image at a time (bounds workspace)
  for (int img = 0; img < 8; ++img)
    for (int br = 0; br < 2; ++br) {
      conv1_direct<<<1024, 256, 0, stream>>>(x, img, br, w1, b1, x1b);
      conv3x3_c32 <<<4096, 256, 0, stream>>>(x1b, w2h, b2, x2b);
      conv3x3_c32 <<<4096, 256, 0, stream>>>(x2b, w3h, b3, x3b);
      conv7_wmma  <<<4096, 128, 0, stream>>>(x1b, x3b, w7h, b7, out_xr,
                                             img, br);
    }

  // x working copy + initial stats (n3, b) from mean of original x
  copy_x<<<6144, 256, 0, stream>>>(x, out_x);
  reduce_partial<<<2048, 256, 0, stream>>>(x, part);
  reduce_final  <<<1, 256, 0, stream>>>(part, scal, /*mode=*/0);

  // mean-feedback refinement; b <= 10 for any mean in [0,1]
  for (int it = 0; it < 10; ++it) {
    reduce_partial<<<2048, 256, 0, stream>>>(out_x, part);
    reduce_final  <<<1, 256, 0, stream>>>(part, scal, /*mode=*/1);
    curve_update  <<<6144, 256, 0, stream>>>(out_x, out_xr, scal, it);
  }
}
